// CGCNN_13194139533623
// MI455X (gfx1250) — matrix operand order repaired and checked by reading, not compiled, not run on hardware
//
#include <hip/hip_runtime.h>
#include <hip/hip_bf16.h>

// ---------------------------------------------------------------------------
// CGCNN forward on MI455X (gfx1250, wave32, WMMA).
// GEMMs run in bf16 with f32 accumulation via V_WMMA_F32_16X16X32_BF16.
// Conv-layer weights are staged into LDS by the Tensor Data Mover
// (TENSOR_LOAD_TO_LDS + s_wait_tensorcnt) when the builtin is available.
// Biases before BatchNorm (train mode) cancel exactly and are skipped.
// ---------------------------------------------------------------------------

typedef __bf16 bf16;
typedef bf16  v16bf __attribute__((ext_vector_type(16)));
typedef bf16  v8bf  __attribute__((ext_vector_type(8)));
typedef float v8f   __attribute__((ext_vector_type(8)));
typedef unsigned int u32x4 __attribute__((ext_vector_type(4)));
typedef int i32x4 __attribute__((ext_vector_type(4)));
typedef int i32x8 __attribute__((ext_vector_type(8)));

#define N_NODES 25000
#define N_EDGES 400000
#define N_GRAPH 128
#define FEAT    64
#define EPS     1e-5f

#if defined(__gfx1250__) && __has_builtin(__builtin_amdgcn_tensor_load_to_lds) && \
    __has_builtin(__builtin_amdgcn_s_wait_tensorcnt)
#define USE_TDM 1
#else
#define USE_TDM 0
#endif

__device__ __forceinline__ v8f wmma_bf16(v16bf a, v16bf b, v8f c) {
  // (neg_a, A, neg_b, B, c_mod, C, reuse_a, reuse_b)
  return __builtin_amdgcn_wmma_f32_16x16x32_bf16(false, a, false, b, (short)0, c,
                                                 false, false);
}

__device__ __forceinline__ v16bf make_afrag(const bf16* p) {
  // 16-bit A 16x32 layout: lane holds K = base..base+7 (VGPR0-3) and
  // K = base+16..base+23 (VGPR4-7): two contiguous 16B loads, 32B apart.
  v8bf lo = *(const v8bf*)(p);
  v8bf hi = *(const v8bf*)(p + 16);
  return __builtin_shufflevector(lo, hi, 0, 1, 2, 3, 4, 5, 6, 7,
                                 8, 9, 10, 11, 12, 13, 14, 15);
}

__device__ __forceinline__ float bn_apply(float x, float sum, float sq,
                                          float invR, float g, float b) {
  float m = sum * invR;
  float v = sq * invR - m * m;
  return g * (x - m) * rsqrtf(v + EPS) + b;
}
__device__ __forceinline__ float sigm(float x) { return 1.f / (1.f + expf(-x)); }

#if USE_TDM
// One TDM descriptor: contiguous 24 KB (6144 dwords) global -> LDS.
// D# group0/group1 per CDNA5 ISA 8.3/8.4: count=1, type=2 ("image"),
// data_size=4B, 1-D tile with tile_dim0 = tensor_dim0 = 6144.
__device__ __forceinline__ void tdm_load_24kb(const bf16* gsrc, bf16* lds_dst) {
  unsigned lds_addr = (unsigned)(unsigned long long)(size_t)lds_dst;  // LDS off
  unsigned long long ga = (unsigned long long)(size_t)gsrc;
  u32x4 g0 = {1u,                                             // count=1
              lds_addr,                                       // lds_addr
              (unsigned)ga,                                   // global[31:0]
              (unsigned)((ga >> 32) & 0x1FFFFFFu) | (2u << 30)};  // type=2
  i32x8 g1 = {(int)(2u << 16),      // data_size = 4 bytes
              (int)(6144u << 16),   // tensor_dim0[15:0] = 6144 (bits 63:48)
              0,                    // tensor_dim0 hi / tensor_dim1 lo
              (int)(6144u << 16),   // tile_dim0 = 6144 (bits 127:112)
              0,                    // tile_dim1/2 unused (1-D)
              6144,                 // tensor_dim0_stride[31:0]
              0, 0};
  i32x4 gz4 = {0, 0, 0, 0};
#if __clang_major__ >= 23
  i32x8 gz8 = {0, 0, 0, 0, 0, 0, 0, 0};
  __builtin_amdgcn_tensor_load_to_lds(g0, g1, gz4, gz4, gz8, 0);
#else
  __builtin_amdgcn_tensor_load_to_lds(g0, g1, gz4, gz4, 0);
#endif
}
#endif

// ---------------------------------------------------------------------------
// Prep: f32 activations -> padded bf16 row-major A
// ---------------------------------------------------------------------------
__global__ void pack_a_k(const float* __restrict__ in, bf16* __restrict__ out,
                         int M, int Kin, int Kpad) {
  long long total = (long long)M * Kpad;
  for (long long idx = (long long)blockIdx.x * 256 + threadIdx.x; idx < total;
       idx += (long long)gridDim.x * 256) {
    int k = (int)(idx % Kpad);
    long long r = idx / Kpad;
    out[idx] = (bf16)((k < Kin) ? in[r * Kin + k] : 0.f);
  }
}

// f32 weight [Kin,64] -> bf16 packed B-fragments (zero-padded K):
// element ((kb*4+nb)*32 + lane)*16 + i  <=  W[kb*32 + ((i&8)<<1) + ((lane&16)>>1) + (i&7)][nb*16 + (lane&15)]
__global__ void pack_w_k(const float* __restrict__ Wsrc, bf16* __restrict__ out,
                         int Kin, int KB) {
  int total = KB * 2048;
  for (int idx = blockIdx.x * 256 + threadIdx.x; idx < total;
       idx += gridDim.x * 256) {
    int i = idx & 15, lane = (idx >> 4) & 31, nbk = idx >> 9;
    int nb = nbk & 3, kb = nbk >> 2;
    int k = kb * 32 + ((i & 8) << 1) + ((lane & 16) >> 1) + (i & 7);
    int n = nb * 16 + (lane & 15);
    out[idx] = (bf16)((k < Kin) ? Wsrc[k * 64 + n] : 0.f);
  }
}

// ---------------------------------------------------------------------------
// Embed GEMM: bf16 A [M,Kpad] @ packed W -> f32 [M,64]. 4 waves/block,
// one 16x64 tile per wave; all A fragments prefetched before the WMMA chain.
// ---------------------------------------------------------------------------
template <int KB>
__global__ __launch_bounds__(128) void gemm_embed_k(
    const bf16* __restrict__ A, int Kpad, const bf16* __restrict__ wpk,
    float* __restrict__ out, int M) {
  int tid = threadIdx.x, wave = tid >> 5, lane = tid & 31;
  int tile = blockIdx.x * 4 + wave;
  int tiles = (M + 15) >> 4;
  if (tile >= tiles) return;
  int row = tile * 16 + (lane & 15);
  if (row >= M) row = M - 1;
  int hi8 = (lane & 16) ? 8 : 0;

  v16bf afr[KB];
#pragma unroll
  for (int kb = 0; kb < KB; ++kb)
    afr[kb] = make_afrag(A + (size_t)row * Kpad + kb * 32 + hi8);

  v8f acc[4] = {};
#pragma unroll
  for (int kb = 0; kb < KB; ++kb) {
#pragma unroll
    for (int nb = 0; nb < 4; ++nb) {
      v16bf b = *(const v16bf*)(wpk + ((kb * 4 + nb) * 32 + lane) * 16);
      acc[nb] = wmma_bf16(afr[kb], b, acc[nb]);
    }
  }
  int colb = lane & 15;
#pragma unroll
  for (int nb = 0; nb < 4; ++nb)
#pragma unroll
    for (int i = 0; i < 8; ++i) {
      int r = tile * 16 + i + hi8;
      if (r < M) out[(size_t)r * 64 + nb * 16 + colb] = acc[nb][i];
    }
}

// ---------------------------------------------------------------------------
// Conv dual GEMM: A = gathered [h_n[src] | h_n[dst] | h_e] (192 cols, bf16),
// two weight sets staged in LDS (TDM when available), produces tmp_m and
// tmp_g f32 [E,64]. 48 WMMAs per wave-tile.
// ---------------------------------------------------------------------------
__global__ __launch_bounds__(128) void conv_gemm_dual_k(
    const int* __restrict__ src, const int* __restrict__ dst,
    const bf16* __restrict__ hN, const bf16* __restrict__ hE,
    const bf16* __restrict__ wm, const bf16* __restrict__ wg,
    float* __restrict__ outM, float* __restrict__ outG, int E) {
  __shared__ bf16 sWm[6 * 2048];
  __shared__ bf16 sWg[6 * 2048];
  int tid = threadIdx.x;
#if USE_TDM
  if (tid < 32) {  // wave 0 issues both DMA descriptors, waits TENSORcnt
    tdm_load_24kb(wm, sWm);
    tdm_load_24kb(wg, sWg);
    __builtin_amdgcn_s_wait_tensorcnt(0);
  }
  __syncthreads();
#else
  {  // cooperative 16B-vector stage of both weight sets (24 KB each)
    const uint4* gm4 = (const uint4*)wm;
    const uint4* gg4 = (const uint4*)wg;
    uint4* sm4 = (uint4*)sWm;
    uint4* sg4 = (uint4*)sWg;
    for (int i = tid; i < 1536; i += 128) { sm4[i] = gm4[i]; sg4[i] = gg4[i]; }
  }
  __syncthreads();
#endif

  int wave = tid >> 5, lane = tid & 31;
  int tile = blockIdx.x * 4 + wave;
  int tiles = (E + 15) >> 4;
  if (tile >= tiles) return;
  int row = tile * 16 + (lane & 15);
  if (row >= E) row = E - 1;
  int si = src[row], di = dst[row];
  int hi8 = (lane & 16) ? 8 : 0;

  const bf16* abase[6];
  abase[0] = hN + (size_t)si * 64;
  abase[1] = hN + (size_t)si * 64 + 32;
  abase[2] = hN + (size_t)di * 64;
  abase[3] = hN + (size_t)di * 64 + 32;
  abase[4] = hE + (size_t)row * 64;
  abase[5] = hE + (size_t)row * 64 + 32;

  // Prefetch all six A fragments (12 b128 loads in flight) so the WMMA
  // chain below overlaps VMEM latency instead of waiting per-fragment.
  v16bf afr[6];
#pragma unroll
  for (int kb = 0; kb < 6; ++kb) afr[kb] = make_afrag(abase[kb] + hi8);

  v8f accM[4] = {};
  v8f accG[4] = {};
#pragma unroll
  for (int kb = 0; kb < 6; ++kb) {
#pragma unroll
    for (int nb = 0; nb < 4; ++nb) {
      v16bf bM = *(const v16bf*)(sWm + ((kb * 4 + nb) * 32 + lane) * 16);
      v16bf bG = *(const v16bf*)(sWg + ((kb * 4 + nb) * 32 + lane) * 16);
      accM[nb] = wmma_bf16(afr[kb], bM, accM[nb]);
      accG[nb] = wmma_bf16(afr[kb], bG, accG[nb]);
    }
  }
  int colb = lane & 15;
#pragma unroll
  for (int nb = 0; nb < 4; ++nb)
#pragma unroll
    for (int i = 0; i < 8; ++i) {
      int r = tile * 16 + i + hi8;
      if (r < E) {
        outM[(size_t)r * 64 + nb * 16 + colb] = accM[nb][i];
        outG[(size_t)r * 64 + nb * 16 + colb] = accG[nb][i];
      }
    }
}

// ---------------------------------------------------------------------------
// Column stats (sum, sumsq) for [R,64] f32; out[0:64]=sum, out[64:128]=sumsq.
// Block partials in LDS, one atomic per column per block.
// ---------------------------------------------------------------------------
__global__ void colstats_k(const float* __restrict__ X, long long total,
                           float* __restrict__ out) {
  __shared__ float ss[256], sq[256];
  int tid = threadIdx.x;
  long long stride = (long long)gridDim.x * 256;  // multiple of 64
  float s = 0.f, q = 0.f;
  for (long long i = (long long)blockIdx.x * 256 + tid; i < total; i += stride) {
    float x = X[i];
    s += x;
    q += x * x;
  }
  ss[tid] = s;
  sq[tid] = q;
  __syncthreads();
  if (tid < 128) { ss[tid] += ss[tid + 128]; sq[tid] += sq[tid + 128]; }
  __syncthreads();
  if (tid < 64) {
    atomicAdd(&out[tid], ss[tid] + ss[tid + 64]);
    atomicAdd(&out[64 + tid], sq[tid] + sq[tid + 64]);
  }
}

// bn + silu; writes f32 (optional) and bf16 copies
__global__ void bn_silu_k(const float* __restrict__ X,
                          const float* __restrict__ st, float invR,
                          const float* __restrict__ g,
                          const float* __restrict__ b, float* __restrict__ outF,
                          bf16* __restrict__ outB, long long total) {
  for (long long idx = (long long)blockIdx.x * 256 + threadIdx.x; idx < total;
       idx += (long long)gridDim.x * 256) {
    int c = (int)(idx & 63);
    float y = bn_apply(X[idx], st[c], st[64 + c], invR, g[c], b[c]);
    float s = y * sigm(y);
    if (outF) outF[idx] = s;
    outB[idx] = (bf16)s;
  }
}

// msg = sigmoid(bn_m) * softplus(bn_g); scatter-add onto agg[dst]
__global__ void msg_scatter_k(
    const float* __restrict__ tm, const float* __restrict__ tg,
    const float* __restrict__ stM, const float* __restrict__ stG,
    const float* __restrict__ gm, const float* __restrict__ bem,
    const float* __restrict__ gg, const float* __restrict__ beg,
    const int* __restrict__ dst, float* __restrict__ agg, float invE,
    long long total) {
  for (long long idx = (long long)blockIdx.x * 256 + threadIdx.x; idx < total;
       idx += (long long)gridDim.x * 256) {
    int c = (int)(idx & 63);
    long long e = idx >> 6;
    float xm = bn_apply(tm[idx], stM[c], stM[64 + c], invE, gm[c], bem[c]);
    float xg = bn_apply(tg[idx], stG[c], stG[64 + c], invE, gg[c], beg[c]);
    float hm = sigm(xm);
    float hg = (xg > 20.f) ? xg : log1pf(expf(xg));
    atomicAdd(&agg[(size_t)dst[e] * 64 + c], hm * hg);
  }
}

// h_n = sigmoid(bn(agg) + h_n); refresh f32 + bf16 copies
__global__ void node_update_k(const float* __restrict__ agg,
                              const float* __restrict__ st, float invN,
                              const float* __restrict__ gn,
                              const float* __restrict__ ben,
                              float* __restrict__ hF, bf16* __restrict__ hB,
                              long long total) {
  for (long long idx = (long long)blockIdx.x * 256 + threadIdx.x; idx < total;
       idx += (long long)gridDim.x * 256) {
    int c = (int)(idx & 63);
    float x = bn_apply(agg[idx], st[c], st[64 + c], invN, gn[c], ben[c]) + hF[idx];
    float y = sigm(x);
    hF[idx] = y;
    hB[idx] = (bf16)y;
  }
}

// segment sums per graph + node counts
__global__ void pool_k(const float* __restrict__ hF, const int* __restrict__ n2g,
                       float* __restrict__ poolS, float* __restrict__ poolC,
                       long long total) {
  for (long long idx = (long long)blockIdx.x * 256 + threadIdx.x; idx < total;
       idx += (long long)gridDim.x * 256) {
    long long node = idx >> 6;
    int c = (int)(idx & 63);
    int gph = n2g[node];
    atomicAdd(&poolS[(size_t)gph * 64 + c], hF[idx]);
    if (c == 0) atomicAdd(&poolC[gph], 1.f);
  }
}

// Head: pooled @ W_fc -> BN(train) -> silu -> @ W_out + b_out. One block,
// thread g owns graph g; cross-graph BN stats via LDS tree reductions.
__global__ __launch_bounds__(128) void head_k(
    const float* __restrict__ poolS, const float* __restrict__ poolC,
    const float* __restrict__ Wfc, const float* __restrict__ gfc,
    const float* __restrict__ befc, const float* __restrict__ Wout,
    const float* __restrict__ bout, float* __restrict__ out) {
  __shared__ float red[128];
  int g = threadIdx.x;
  float cnt = fmaxf(poolC[g], 1.f);
  float v[10];
#pragma unroll
  for (int j = 0; j < 10; ++j) v[j] = 0.f;  // b_fc cancels in BN
  for (int c = 0; c < 64; ++c) {
    float p = poolS[(size_t)g * 64 + c] / cnt;
#pragma unroll
    for (int j = 0; j < 10; ++j) v[j] += p * Wfc[c * 10 + j];
  }
  float o = bout[0];
  for (int j = 0; j < 10; ++j) {
    red[g] = v[j];
    __syncthreads();
    for (int off = 64; off > 0; off >>= 1) {
      if (g < off) red[g] += red[g + off];
      __syncthreads();
    }
    float m = red[0] / 128.f;
    __syncthreads();
    red[g] = v[j] * v[j];
    __syncthreads();
    for (int off = 64; off > 0; off >>= 1) {
      if (g < off) red[g] += red[g + off];
      __syncthreads();
    }
    float q = red[0] / 128.f;
    __syncthreads();
    float y = gfc[j] * (v[j] - m) * rsqrtf(q - m * m + EPS) + befc[j];
    y = y * sigm(y);
    o += y * Wout[j];
  }
  out[g] = o;
}

// ---------------------------------------------------------------------------
// Host orchestration
// ---------------------------------------------------------------------------
static inline int gsblocks(long long total) {
  long long b = (total + 255) / 256;
  return (int)(b < 4096 ? b : 4096);
}

extern "C" void kernel_launch(void* const* d_in, const int* in_sizes, int n_in,
                              void* d_out, int out_size, void* d_ws,
                              size_t ws_size, hipStream_t stream) {
  const float* node_feats = (const float*)d_in[0];
  const float* edge_feats = (const float*)d_in[1];
  const int* src = (const int*)d_in[2];
  const int* dst = (const int*)d_in[3];
  const int* n2g = (const int*)d_in[4];
  const float* W_ne = (const float*)d_in[5];
  const float* g_ne = (const float*)d_in[7];
  const float* be_ne = (const float*)d_in[8];
  const float* W_ee = (const float*)d_in[9];
  const float* g_ee = (const float*)d_in[11];
  const float* be_ee = (const float*)d_in[12];
  const float* Wm = (const float*)d_in[13];
  const float* gm = (const float*)d_in[15];
  const float* bem = (const float*)d_in[16];
  const float* Wg = (const float*)d_in[17];
  const float* gg = (const float*)d_in[19];
  const float* beg = (const float*)d_in[20];
  const float* gn = (const float*)d_in[21];
  const float* ben = (const float*)d_in[22];
  const float* W_fc = (const float*)d_in[23];
  const float* g_fc = (const float*)d_in[25];
  const float* be_fc = (const float*)d_in[26];
  const float* W_out = (const float*)d_in[27];
  const float* b_out = (const float*)d_in[28];
  float* out = (float*)d_out;

  char* base = (char*)d_ws;
  size_t off = 0;
  auto alloc = [&](size_t bytes) -> char* {
    off = (off + 255) & ~(size_t)255;
    char* p = base + off;
    off += bytes;
    return p;
  };
  bf16* aNode = (bf16*)alloc((size_t)N_NODES * 96 * 2);
  bf16* aEdge = (bf16*)alloc((size_t)N_EDGES * 64 * 2);
  bf16* wpk_ne = (bf16*)alloc(3 * 2048 * 2);
  bf16* wpk_ee = (bf16*)alloc(2 * 2048 * 2);
  bf16* wpk_m = (bf16*)alloc(3 * 6 * 2048 * 2);
  bf16* wpk_g = (bf16*)alloc(3 * 6 * 2048 * 2);
  float* tmp_n = (float*)alloc((size_t)N_NODES * 64 * 4);
  float* h_nF = (float*)alloc((size_t)N_NODES * 64 * 4);
  bf16* h_nB = (bf16*)alloc((size_t)N_NODES * 64 * 2);
  bf16* h_eB = (bf16*)alloc((size_t)N_EDGES * 64 * 2);
  float* tmp_m = (float*)alloc((size_t)N_EDGES * 64 * 4);  // also edge-embed tmp
  float* tmp_g = (float*)alloc((size_t)N_EDGES * 64 * 4);
  float* agg = (float*)alloc((size_t)N_NODES * 64 * 4);
  float* stM = (float*)alloc(128 * 4);
  float* stG = (float*)alloc(128 * 4);
  float* poolS = (float*)alloc((size_t)N_GRAPH * 64 * 4);
  float* poolC = (float*)alloc((size_t)N_GRAPH * 4);

  const long long nTot = (long long)N_NODES * 64;
  const long long eTot = (long long)N_EDGES * 64;
  const float invN = 1.f / N_NODES, invE = 1.f / N_EDGES;

  // --- pack inputs & weights ---
  pack_a_k<<<gsblocks((long long)N_NODES * 96), 256, 0, stream>>>(
      node_feats, aNode, N_NODES, 92, 96);
  pack_a_k<<<gsblocks((long long)N_EDGES * 64), 256, 0, stream>>>(
      edge_feats, aEdge, N_EDGES, 41, 64);
  pack_w_k<<<24, 256, 0, stream>>>(W_ne, wpk_ne, 92, 3);
  pack_w_k<<<16, 256, 0, stream>>>(W_ee, wpk_ee, 41, 2);
  for (int l = 0; l < 3; ++l) {
    pack_w_k<<<48, 256, 0, stream>>>(Wm + (size_t)l * 192 * 64,
                                     wpk_m + (size_t)l * 12288, 192, 6);
    pack_w_k<<<48, 256, 0, stream>>>(Wg + (size_t)l * 192 * 64,
                                     wpk_g + (size_t)l * 12288, 192, 6);
  }

  // --- node embedding ---
  {
    int tiles = (N_NODES + 15) / 16;
    gemm_embed_k<3><<<(tiles + 3) / 4, 128, 0, stream>>>(aNode, 96, wpk_ne,
                                                         tmp_n, N_NODES);
    hipMemsetAsync(stM, 0, 128 * 4, stream);
    colstats_k<<<512, 256, 0, stream>>>(tmp_n, nTot, stM);
    bn_silu_k<<<gsblocks(nTot), 256, 0, stream>>>(tmp_n, stM, invN, g_ne, be_ne,
                                                  h_nF, h_nB, nTot);
  }
  // --- edge embedding ---
  {
    int tiles = (N_EDGES + 15) / 16;
    gemm_embed_k<2><<<(tiles + 3) / 4, 128, 0, stream>>>(aEdge, 64, wpk_ee,
                                                         tmp_m, N_EDGES);
    hipMemsetAsync(stM, 0, 128 * 4, stream);
    colstats_k<<<512, 256, 0, stream>>>(tmp_m, eTot, stM);
    bn_silu_k<<<gsblocks(eTot), 256, 0, stream>>>(tmp_m, stM, invE, g_ee, be_ee,
                                                  (float*)nullptr, h_eB, eTot);
  }
  // --- conv layers ---
  for (int l = 0; l < 3; ++l) {
    int tiles = (N_EDGES + 15) / 16;
    conv_gemm_dual_k<<<(tiles + 3) / 4, 128, 0, stream>>>(
        src, dst, h_nB, h_eB, wpk_m + (size_t)l * 12288,
        wpk_g + (size_t)l * 12288, tmp_m, tmp_g, N_EDGES);
    hipMemsetAsync(stM, 0, 128 * 4, stream);
    hipMemsetAsync(stG, 0, 128 * 4, stream);
    colstats_k<<<512, 256, 0, stream>>>(tmp_m, eTot, stM);
    colstats_k<<<512, 256, 0, stream>>>(tmp_g, eTot, stG);
    hipMemsetAsync(agg, 0, (size_t)N_NODES * 64 * 4, stream);
    msg_scatter_k<<<gsblocks(eTot), 256, 0, stream>>>(
        tmp_m, tmp_g, stM, stG, gm + l * 64, bem + l * 64, gg + l * 64,
        beg + l * 64, dst, agg, invE, eTot);
    hipMemsetAsync(stM, 0, 128 * 4, stream);
    colstats_k<<<512, 256, 0, stream>>>(agg, nTot, stM);
    node_update_k<<<gsblocks(nTot), 256, 0, stream>>>(
        agg, stM, invN, gn + l * 64, ben + l * 64, h_nF, h_nB, nTot);
  }
  // --- pooling + head ---
  hipMemsetAsync(poolS, 0, (size_t)N_GRAPH * 64 * 4, stream);
  hipMemsetAsync(poolC, 0, (size_t)N_GRAPH * 4, stream);
  pool_k<<<gsblocks(nTot), 256, 0, stream>>>(h_nF, n2g, poolS, poolC, nTot);
  head_k<<<1, 128, 0, stream>>>(poolS, poolC, W_fc, g_fc, be_fc, W_out, b_out,
                                out);
}
